// HelloNPUModel_76879914598602
// MI455X (gfx1250) — matrix operand order repaired and checked by reading, not compiled, not run on hardware
//
#include <hip/hip_runtime.h>

#define SEQ 8192

typedef __attribute__((ext_vector_type(16))) _Float16 v16h;
typedef __attribute__((ext_vector_type(8)))  _Float16 h8;
typedef __attribute__((ext_vector_type(2)))  _Float16 h2;
typedef __attribute__((ext_vector_type(2)))  __fp16   fp16x2;
typedef __attribute__((ext_vector_type(8)))  float    v8f;

#if __has_builtin(__builtin_amdgcn_exp2f)
__device__ __forceinline__ float fast_exp2(float x) {
  return __builtin_amdgcn_exp2f(x);  // bare v_exp_f32, no denorm fixup
}
#else
__device__ __forceinline__ float fast_exp2(float x) {
  return __expf(0.69314718056f * x);
}
#endif

__device__ __forceinline__ v8f wmma_f16(v16h a, v16h b, v8f c) {
  // D = A(16x32 f16) * B(32x16 f16) + C(16x16 f32)
  return __builtin_amdgcn_wmma_f32_16x16x32_f16(false, a, false, b, (short)0, c,
                                                false, false);
}

__device__ __forceinline__ h2 pk_f16(float x, float y) {
  const fp16x2 t = __builtin_amdgcn_cvt_pkrtz(x, y);
  return __builtin_bit_cast(h2, t);
}

__device__ __forceinline__ v16h zero_v16h() {
  v16h z;
#pragma unroll
  for (int i = 0; i < 16; ++i) z[i] = (_Float16)0.0f;
  return z;
}

// A-operand with real K=0..15 only (K=16..31 zero).  Per ISA 16-bit A layout:
// lanes 0-15 hold K 0-7 (lo) / 16-23 (hi); lanes 16-31 hold K 8-15 / 24-31.
__device__ __forceinline__ v16h load_a_pad16(const _Float16* row, int lane) {
  const h8 lo = *reinterpret_cast<const h8*>(row + ((lane & 16) ? 8 : 0));
  v16h a;
#pragma unroll
  for (int i = 0; i < 8; ++i) { a[i] = lo[i]; a[i + 8] = (_Float16)0.0f; }
  return a;
}

// Full K=32 A-operand; `row` points at 32 contiguous halves of one matrix row.
__device__ __forceinline__ v16h load_a_k32(const _Float16* row, int lane) {
  const int b0 = (lane & 16) ? 8 : 0;
  const h8 lo = *reinterpret_cast<const h8*>(row + b0);
  const h8 hi = *reinterpret_cast<const h8*>(row + b0 + 16);
  v16h a;
#pragma unroll
  for (int i = 0; i < 8; ++i) { a[i] = lo[i]; a[i + 8] = hi[i]; }
  return a;
}

// ---------------------------------------------------------------------------
// Kernel A: qkv = src @ w_qkv^T  + fp16 staging copies.
// qkv_h is pre-scaled by 0.5 so S^T = (0.5K)(0.5Q)^T = (Q K^T)/4 directly.
// qkvT_h (the V operand) stays unscaled.
// ---------------------------------------------------------------------------
__global__ __launch_bounds__(256) void prep_kernel(
    const float* __restrict__ src, const float* __restrict__ w_qkv,
    const float* __restrict__ w_ffn1, const float* __restrict__ w_ffn2,
    _Float16* __restrict__ qkv_h, _Float16* __restrict__ qkvT_h,
    _Float16* __restrict__ srcT_h, _Float16* __restrict__ w1_h,
    _Float16* __restrict__ w2_h, _Float16* __restrict__ w2t_h) {
  const int id = blockIdx.x * blockDim.x + threadIdx.x;  // 0 .. SEQ*16-1
  const int s = id >> 4;
  const int d = id & 15;
  float a = 0.0f;
#pragma unroll
  for (int k = 0; k < 16; ++k) a += src[s * 16 + k] * w_qkv[d * 16 + k];
  qkv_h[s * 16 + d] = (_Float16)(0.5f * a);  // pre-scaled Q/K copy
  qkvT_h[d * SEQ + s] = (_Float16)a;         // unscaled V (col-major)
  srcT_h[d * SEQ + s] = (_Float16)src[s * 16 + d];
  if (id < 1024) {
    w1_h[id] = (_Float16)w_ffn1[id];  // [64][16] row-major
    w2_h[id] = (_Float16)w_ffn2[id];  // [16][64] row-major
    const int hh = id >> 4, dd = id & 15;
    w2t_h[id] = (_Float16)w_ffn2[dd * 64 + hh];  // [64][16] = w_ffn2^T
  }
}

// ---------------------------------------------------------------------------
// Kernel B: flash attention (transposed scores, lazy-rescale softmax) + FFN
// fwd + loss grad + grad_ffn_h.  One wave per 16 query rows.
// ---------------------------------------------------------------------------
__global__ __launch_bounds__(128) void flash_ffn_kernel(
    const float* __restrict__ tgt, const _Float16* __restrict__ qkv_h,
    const _Float16* __restrict__ qkvT_h, const _Float16* __restrict__ w1_h,
    const _Float16* __restrict__ w2_h, const _Float16* __restrict__ w2t_h,
    float* __restrict__ out_pred, _Float16* __restrict__ lgT_h,
    _Float16* __restrict__ fhT_h, _Float16* __restrict__ gfhT_h,
    _Float16* __restrict__ ctxT_h) {
  __shared__ _Float16 lds_t[4][16][16];   // per-wave 16x16 transpose buffer
  __shared__ _Float16 lds_fh[4][16][64];  // per-wave ffn_h transpose buffer

  const int lane = threadIdx.x & 31;
  const int warp = threadIdx.x >> 5;
  const int q0 = (blockIdx.x * 4 + warp) * 16;
  const int row16 = lane & 15;
  const int half = lane >> 4;  // which 16-lane half

  _Float16(*buft)[16] = lds_t[warp];
  _Float16(*buffh)[64] = lds_fh[warp];

  // B-operand of K@Q^T: B^T rows are (pre-scaled) query rows (K = D, padded)
  v16h b_q = load_a_pad16(qkv_h + (q0 + row16) * 16, lane);

  v8f acc = {};          // context accumulator (C layout: row=query)
  float m_run = -1e30f;  // per-query reference max (wave-consistent)
  float l_run = 0.0f;    // this half's partial sum

  const float LOG2E = 1.442695041f;

  for (int k0 = 0; k0 < SEQ; k0 += 32) {
    const v16h a_k0 = load_a_pad16(qkv_h + (k0 + row16) * 16, lane);
    const v16h a_k1 = load_a_pad16(qkv_h + (k0 + 16 + row16) * 16, lane);
    v8f zc = {};
    // s0[r] = S^T[key k0 + r + 8*half][query q0+row16] (already /4)
    v8f s0 = wmma_f16(a_k0, b_q, zc);
    v8f s1 = wmma_f16(a_k1, b_q, zc);

    float mloc = -1e30f;
#pragma unroll
    for (int r = 0; r < 8; ++r) mloc = fmaxf(mloc, fmaxf(s0[r], s1[r]));

    // Lazy rescale: exp(s - m_run) stays <= e^10 < fp16 max, so only when a
    // lane's block-max exceeds the margin do we merge maxima and rescale.
    if (__builtin_expect(__ballot(mloc > m_run + 10.0f) != 0, 0)) {
      const float mall = fmaxf(mloc, __shfl_xor(mloc, 16, 32));
      const float mn = fmaxf(m_run, mall);
      const float alpha = __expf(m_run - mn);
      m_run = mn;
      l_run *= alpha;
#pragma unroll
      for (int r = 0; r < 8; ++r) acc[r] *= __shfl(alpha, r + 8 * half, 32);
    }

    const float mb = m_run * LOG2E;
    float p0[8], p1[8];
    float lsum = 0.0f;
#pragma unroll
    for (int r = 0; r < 8; ++r) {
      p0[r] = fast_exp2(fmaf(s0[r], LOG2E, -mb));
      p1[r] = fast_exp2(fmaf(s1[r], LOG2E, -mb));
      lsum += p0[r] + p1[r];
    }
    l_run += lsum;

    // P registers already sit in the A-operand layout for P@V; pack-convert.
    union { v16h v; h2 q[8]; } ap;
#pragma unroll
    for (int j = 0; j < 4; ++j) {
      ap.q[j]     = pk_f16(p0[2 * j], p0[2 * j + 1]);
      ap.q[j + 4] = pk_f16(p1[2 * j], p1[2 * j + 1]);
    }
    // V block as B-operand: qkv^T rows (col-major V), full K=32
    const v16h b_v =
        load_a_k32(qkvT_h + row16 * SEQ + k0, lane);
    acc = wmma_f16(ap.v, b_v, acc);
  }

  // finish softmax: combine the two key-halves' sums, normalize context
  l_run += __shfl_xor(l_run, 16, 32);
#pragma unroll
  for (int r = 0; r < 8; ++r) {
    const float lr = __shfl(l_run, r + 8 * half, 32);
    acc[r] *= (1.0f / lr);
  }

  // ctx^T fp16 (16B store per lane: rows q0+8*half .. +7 are contiguous)
  {
    h8 v;
#pragma unroll
    for (int r = 0; r < 8; ++r) v[r] = (_Float16)acc[r];
    *reinterpret_cast<h8*>(ctxT_h + row16 * SEQ + q0 + 8 * half) = v;
  }
  // context -> A layout (K=16 pad) via LDS transpose (once per tile)
#pragma unroll
  for (int r = 0; r < 8; ++r) buft[r + 8 * half][row16] = (_Float16)acc[r];
  asm volatile("s_wait_dscnt 0" ::: "memory");
  const v16h a_ctx = load_a_pad16(&buft[row16][0], lane);

  // ffn_h = relu(ctx @ w_ffn1^T): B-operand lanes are w_ffn1 rows (K=D pad)
  v8f fh[4];
#pragma unroll
  for (int j = 0; j < 4; ++j) {
    v16h b_w1;
    b_w1 = load_a_pad16(w1_h + (16 * j + row16) * 16, lane);
    v8f zc = {};
    fh[j] = wmma_f16(a_ctx, b_w1, zc);
#pragma unroll
    for (int r = 0; r < 8; ++r) fh[j][r] = fmaxf(fh[j][r], 0.0f);
    h8 v;
#pragma unroll
    for (int r = 0; r < 8; ++r) v[r] = (_Float16)fh[j][r];
    *reinterpret_cast<h8*>(fhT_h + (16 * j + row16) * SEQ + q0 + 8 * half) = v;
#pragma unroll
    for (int r = 0; r < 8; ++r)
      buffh[r + 8 * half][16 * j + row16] = (_Float16)fh[j][r];
  }
  asm volatile("s_wait_dscnt 0" ::: "memory");

  // prediction = ffn_h @ w_ffn2^T  (K = H = 64, two full-K WMMAs)
  const v16h a_fh0 = load_a_k32(&buffh[row16][0], lane);
  const v16h a_fh1 = load_a_k32(&buffh[row16][32], lane);
  const v16h b_w2a =
      load_a_k32(w2_h + row16 * 64, lane);
  const v16h b_w2b =
      load_a_k32(w2_h + row16 * 64 + 32, lane);
  v8f pred = {};
  pred = wmma_f16(a_fh0, b_w2a, pred);
  pred = wmma_f16(a_fh1, b_w2b, pred);

  // loss grad, prediction output
  float lg[8];
#pragma unroll
  for (int r = 0; r < 8; ++r) {
    const int srow = q0 + r + 8 * half;
    const float p = pred[r];
    out_pred[srow * 16 + row16] = p;
    lg[r] = p - tgt[srow * 16 + row16];
  }
  {
    h8 v;
#pragma unroll
    for (int r = 0; r < 8; ++r) v[r] = (_Float16)lg[r];
    *reinterpret_cast<h8*>(lgT_h + row16 * SEQ + q0 + 8 * half) = v;
  }
  // lg -> A layout via LDS transpose
#pragma unroll
  for (int r = 0; r < 8; ++r) buft[r + 8 * half][row16] = (_Float16)lg[r];
  asm volatile("s_wait_dscnt 0" ::: "memory");
  const v16h a_lg = load_a_pad16(&buft[row16][0], lane);

  // grad_ffn_h = relu-mask(lg @ w_ffn2): B^T rows come from w_ffn2^T copy
#pragma unroll
  for (int j = 0; j < 4; ++j) {
    v16h b_w2t;
    b_w2t = load_a_pad16(w2t_h + (16 * j + row16) * 16, lane);
    v8f zc = {};
    v8f g = wmma_f16(a_lg, b_w2t, zc);
#pragma unroll
    for (int r = 0; r < 8; ++r) g[r] = (fh[j][r] <= 0.0f) ? 0.0f : g[r];
    h8 v;
#pragma unroll
    for (int r = 0; r < 8; ++r) v[r] = (_Float16)g[r];
    *reinterpret_cast<h8*>(gfhT_h + (16 * j + row16) * SEQ + q0 + 8 * half) = v;
  }
}

// ---------------------------------------------------------------------------
// Kernel C: weight grads  X^T @ Y  over S=8192.  9 output tiles, 8-wave
// split-K per tile, deterministic LDS reduction (no float atomics).
// ---------------------------------------------------------------------------
__global__ __launch_bounds__(256) void wgrad_kernel(
    const _Float16* __restrict__ lgT_h, const _Float16* __restrict__ srcT_h,
    const _Float16* __restrict__ fhT_h, const _Float16* __restrict__ gfhT_h,
    const _Float16* __restrict__ ctxT_h, float* __restrict__ out) {
  __shared__ float red[8][256];
  const int lane = threadIdx.x & 31;
  const int warp = threadIdx.x >> 5;
  const int row16 = lane & 15;
  const int half = lane >> 4;
  const int t = blockIdx.x;  // 0: gw_qkv, 1-4: gw_ffn2 tiles, 5-8: gw_ffn1

  const _Float16* Abase;
  const _Float16* Bbase;
  if (t == 0) {
    Abase = lgT_h + row16 * SEQ;   // A = lg^T rows (d)
    Bbase = srcT_h + row16 * SEQ;  // B^T = src^T rows (d)
  } else if (t <= 4) {
    Abase = lgT_h + row16 * SEQ;   // gw_ffn2 = lg^T @ fh
    Bbase = fhT_h + (16 * (t - 1) + row16) * SEQ;
  } else {
    Abase = gfhT_h + (16 * (t - 5) + row16) * SEQ;  // gw_ffn1 = gfh^T @ ctx
    Bbase = ctxT_h + row16 * SEQ;
  }

  v8f acc = {};
  for (int kb = warp; kb < SEQ / 32; kb += 8) {
    const int k0 = kb * 32;
    const v16h a = load_a_k32(Abase + k0, lane);
    const v16h b = load_a_k32(Bbase + k0, lane);
    acc = wmma_f16(a, b, acc);
  }
#pragma unroll
  for (int r = 0; r < 8; ++r)
    red[warp][(r + 8 * half) * 16 + row16] = acc[r];
  __syncthreads();

  const int e = threadIdx.x;
  float sum = 0.0f;
#pragma unroll
  for (int w = 0; w < 8; ++w) sum += red[w][e];
  const int M = e >> 4, N = e & 15;
  // out layout: pred[8192*16], gw_qkv[16*16], gw_ffn1[64*16], gw_ffn2[16*64]
  if (t == 0)
    out[SEQ * 16 + M * 16 + N] = sum;
  else if (t <= 4)
    out[SEQ * 16 + 256 + 1024 + M * 64 + 16 * (t - 1) + N] = sum;
  else
    out[SEQ * 16 + 256 + (16 * (t - 5) + M) * 16 + N] = sum;
}

// ---------------------------------------------------------------------------
extern "C" void kernel_launch(void* const* d_in, const int* in_sizes, int n_in,
                              void* d_out, int out_size, void* d_ws,
                              size_t ws_size, hipStream_t stream) {
  const float* src = (const float*)d_in[0];
  const float* tgt = (const float*)d_in[1];
  const float* w_qkv = (const float*)d_in[2];
  const float* w_ffn1 = (const float*)d_in[3];
  const float* w_ffn2 = (const float*)d_in[4];
  float* out = (float*)d_out;

  _Float16* ws = (_Float16*)d_ws;          // ~3.3 MB total
  _Float16* qkv_h = ws;                    // [8192][16] (x0.5)
  _Float16* qkvT_h = qkv_h + SEQ * 16;     // [16][8192]
  _Float16* srcT_h = qkvT_h + SEQ * 16;    // [16][8192]
  _Float16* lgT_h = srcT_h + SEQ * 16;     // [16][8192]
  _Float16* ctxT_h = lgT_h + SEQ * 16;     // [16][8192]
  _Float16* fhT_h = ctxT_h + SEQ * 16;     // [64][8192]
  _Float16* gfhT_h = fhT_h + SEQ * 64;     // [64][8192]
  _Float16* w1_h = gfhT_h + SEQ * 64;      // [64][16]
  _Float16* w2_h = w1_h + 1024;            // [16][64]
  _Float16* w2t_h = w2_h + 1024;           // [64][16]

  prep_kernel<<<512, 256, 0, stream>>>(src, w_qkv, w_ffn1, w_ffn2, qkv_h,
                                       qkvT_h, srcT_h, w1_h, w2_h, w2t_h);
  flash_ffn_kernel<<<128, 128, 0, stream>>>(tgt, qkv_h, qkvT_h, w1_h, w2_h,
                                            w2t_h, out, lgT_h, fhT_h, gfhT_h,
                                            ctxT_h);
  wgrad_kernel<<<9, 256, 0, stream>>>(lgT_h, srcT_h, fhT_h, gfhT_h, ctxT_h,
                                      out);
}
